// Prototype_Pool_27135603376736
// MI455X (gfx1250) — matrix operand order repaired and checked by reading, not compiled, not run on hardware
//
#include <hip/hip_runtime.h>

typedef __attribute__((ext_vector_type(16))) _Float16 v16h;
typedef __attribute__((ext_vector_type(4)))  _Float16 v4h;
typedef __attribute__((ext_vector_type(8)))  float    v8f;
typedef __attribute__((ext_vector_type(4)))  float    v4f;

#define K_DIM    1024
#define NMEM_PAD 48   // 40 real memory rows padded to 3 WMMA N-tiles

// ---------------------------------------------------------------------------
// Prep: mn_h[row][k] = fp16( memory[row][k] / max(||memory[row]||, 1e-8) )
// Rows >= nmem are zero-filled (padding tiles; masked out of the argmax).
// One block per padded row; 256 threads x 4 elements.
// ---------------------------------------------------------------------------
__global__ __launch_bounds__(256) void prep_memory_kernel(
    const float* __restrict__ mem, _Float16* __restrict__ mn_h, int nmem) {
  const int row = blockIdx.x;
  const int t   = threadIdx.x;
  __shared__ float red[256];

  if (row >= nmem) {                       // uniform per block: no divergence
    v4h z = {};
    *(v4h*)(mn_h + (size_t)row * K_DIM + t * 4) = z;
    return;
  }

  v4f v = *(const v4f*)(mem + (size_t)row * K_DIM + t * 4);
  red[t] = v.x * v.x + v.y * v.y + v.z * v.z + v.w * v.w;
  __syncthreads();
  for (int off = 128; off > 0; off >>= 1) {
    if (t < off) red[t] += red[t + off];
    __syncthreads();
  }
  const float inv = 1.0f / fmaxf(sqrtf(red[0]), 1e-8f);

  v4h h;
  h.x = (_Float16)(v.x * inv);
  h.y = (_Float16)(v.y * inv);
  h.z = (_Float16)(v.z * inv);
  h.w = (_Float16)(v.w * inv);
  *(v4h*)(mn_h + (size_t)row * K_DIM + t * 4) = h;
}

// ---------------------------------------------------------------------------
// Main: per wave, 16 rows of x. WMMA f32<-f16 16x16x32, 3 N-tiles (48 cols),
// K swept 1024 in steps of 32. The streamed x data is also written out as the
// second output (x copy) with non-temporal stores -> x read from HBM once.
// Then LDS spill of the 16x48 sims tile, per-row argmax over n<40, and a
// 32-lane gather-copy of the winning memory row (L2-resident source).
// ---------------------------------------------------------------------------
__global__ __launch_bounds__(256) void simtop_kernel(
    const float* __restrict__ x, const float* __restrict__ mem,
    const _Float16* __restrict__ mn_h, float* __restrict__ out_near,
    float* __restrict__ out_x, int nmem) {
  __shared__ float sims[8][16][NMEM_PAD];
  __shared__ int   best[8][16];

  const int lane = threadIdx.x & 31;
  const int wave = threadIdx.x >> 5;
  const int r16  = lane & 15;
  const int hi   = lane >> 4;             // lane half (0: lanes 0-15, 1: 16-31)
  const size_t base = ((size_t)blockIdx.x * 8 + wave) * 16;
  const size_t row  = base + r16;

  const float* xr = x     + row * K_DIM;
  float*       ox = out_x + row * K_DIM;
  const int ko = hi * 8;                  // A-layout K sub-offset per lane half

  v8f acc0 = {}, acc1 = {}, acc2 = {};

  for (int kb = 0; kb < K_DIM; kb += 32) {
    __builtin_prefetch(xr + kb + 256, 0, 0);   // stream-ahead on x

    // A fragment source (and x-copy payload): 2x8 contiguous floats per lane.
    v4f f0 = *(const v4f*)(xr + kb + ko);
    v4f f1 = *(const v4f*)(xr + kb + ko + 4);
    v4f f2 = *(const v4f*)(xr + kb + 16 + ko);
    v4f f3 = *(const v4f*)(xr + kb + 16 + ko + 4);

    // Second output: exact copy of x, streamed through without a second read.
    __builtin_nontemporal_store(f0, (v4f*)(ox + kb + ko));
    __builtin_nontemporal_store(f1, (v4f*)(ox + kb + ko + 4));
    __builtin_nontemporal_store(f2, (v4f*)(ox + kb + 16 + ko));
    __builtin_nontemporal_store(f3, (v4f*)(ox + kb + 16 + ko + 4));

    // 16-bit A 16x32 layout: halves 0-7 = K(kb+ko .. +8), 8-15 = K(kb+16+ko .. +8)
    v16h a;
    a[0]  = (_Float16)f0.x; a[1]  = (_Float16)f0.y;
    a[2]  = (_Float16)f0.z; a[3]  = (_Float16)f0.w;
    a[4]  = (_Float16)f1.x; a[5]  = (_Float16)f1.y;
    a[6]  = (_Float16)f1.z; a[7]  = (_Float16)f1.w;
    a[8]  = (_Float16)f2.x; a[9]  = (_Float16)f2.y;
    a[10] = (_Float16)f2.z; a[11] = (_Float16)f2.w;
    a[12] = (_Float16)f3.x; a[13] = (_Float16)f3.y;
    a[14] = (_Float16)f3.z; a[15] = (_Float16)f3.w;

    // B 32x16 layout: lane -> column n = t*16 + r16, halves 0-7 = K(kb+hi*8 .. +8),
    // 8-15 = K(kb+16+hi*8 .. +8); two 16B loads from row n of mn_h.
    typedef __attribute__((ext_vector_type(8))) _Float16 v8h_t;
    const int kbB = kb + hi * 8;
    const _Float16* p0 = mn_h + (size_t)(r16     ) * K_DIM + kbB;
    const _Float16* p1 = mn_h + (size_t)(r16 + 16) * K_DIM + kbB;
    const _Float16* p2 = mn_h + (size_t)(r16 + 32) * K_DIM + kbB;
    v16h b0 = __builtin_shufflevector(*(const v8h_t*)p0, *(const v8h_t*)(p0 + 16), 0,1,2,3,4,5,6,7,8,9,10,11,12,13,14,15);
    v16h b1 = __builtin_shufflevector(*(const v8h_t*)p1, *(const v8h_t*)(p1 + 16), 0,1,2,3,4,5,6,7,8,9,10,11,12,13,14,15);
    v16h b2 = __builtin_shufflevector(*(const v8h_t*)p2, *(const v8h_t*)(p2 + 16), 0,1,2,3,4,5,6,7,8,9,10,11,12,13,14,15);

    acc0 = __builtin_amdgcn_wmma_f32_16x16x32_f16(false, a, false, b0,
                                                  (short)0, acc0, false, false);
    acc1 = __builtin_amdgcn_wmma_f32_16x16x32_f16(false, a, false, b1,
                                                  (short)0, acc1, false, false);
    acc2 = __builtin_amdgcn_wmma_f32_16x16x32_f16(false, a, false, b2,
                                                  (short)0, acc2, false, false);
  }

  // D layout: VGPR v holds row (v + 8*hi), column r16 (+16 per N-tile).
  #pragma unroll
  for (int v = 0; v < 8; ++v) {
    const int m = v + hi * 8;
    sims[wave][m][r16     ] = acc0[v];
    sims[wave][m][r16 + 16] = acc1[v];
    sims[wave][m][r16 + 32] = acc2[v];
  }
  __syncthreads();

  // Per-row argmax over the 40 real columns; '>' keeps first index on ties,
  // matching jax.lax.top_k ordering.
  if (lane < 16) {
    float bv = sims[wave][lane][0];
    int   bi = 0;
    for (int j = 1; j < nmem; ++j) {
      const float s = sims[wave][lane][j];
      if (s > bv) { bv = s; bi = j; }
    }
    best[wave][lane] = bi;
  }
  __syncthreads();

  // Gather-copy the winning ORIGINAL memory row (reference indexes the
  // un-normalized memory). 160 KB source sits in L2; stores are NT.
  for (int r = 0; r < 16; ++r) {
    const int bi = best[wave][r];
    const v4f* src = (const v4f*)(mem + (size_t)bi * K_DIM);
    v4f*       dst = (v4f*)(out_near + (base + r) * K_DIM);
    #pragma unroll
    for (int i = lane; i < K_DIM / 4; i += 32)
      __builtin_nontemporal_store(src[i], dst + i);
  }
}

// ---------------------------------------------------------------------------
extern "C" void kernel_launch(void* const* d_in, const int* in_sizes, int n_in,
                              void* d_out, int out_size, void* d_ws,
                              size_t ws_size, hipStream_t stream) {
  const float* x   = (const float*)d_in[0];
  const float* mem = (const float*)d_in[1];
  // d_in[2] = top_k (only idx[:,0] is used by the reference -> argmax)

  const int nrows = in_sizes[0] / K_DIM;   // 65536
  const int nmem  = in_sizes[1] / K_DIM;   // 40

  _Float16* mn_h = (_Float16*)d_ws;        // 48*1024*2 = 96 KB scratch

  float* out_near = (float*)d_out;
  float* out_x    = (float*)d_out + (size_t)nrows * K_DIM;

  prep_memory_kernel<<<NMEM_PAD, 256, 0, stream>>>(mem, mn_h, nmem);

  // 8 waves/block * 16 rows/wave = 128 rows per block
  simtop_kernel<<<nrows / 128, 256, 0, stream>>>(x, mem, mn_h, out_near,
                                                 out_x, nmem);
}
